// Model_47167330845004
// MI455X (gfx1250) — matrix operand order repaired and checked by reading, not compiled, not run on hardware
//
#include <hip/hip_runtime.h>
#include <hip/hip_bf16.h>

typedef __attribute__((ext_vector_type(16))) _Float16 v16h;
typedef __attribute__((ext_vector_type(8)))  float    v8f;
typedef int v4i_vs __attribute__((vector_size(16)));   // matches builtin's V4i param

#define BATCH  512
#define SEQ    112
#define DMODEL 128
#define NHEADS 8
#define HD     16
#define KCONV  8

// ---- CDNA5 async global->LDS path (ASYNCcnt), with sync fallback -------------
#if __has_builtin(__builtin_amdgcn_global_load_async_to_lds_b128) && \
    __has_builtin(__builtin_amdgcn_global_load_async_to_lds_b32)
#define HAS_ASYNC_LDS 1
#else
#define HAS_ASYNC_LDS 0
#endif

#define GAS128(p) ((__attribute__((address_space(1))) v4i_vs*)(p))
#define LAS128(p) ((__attribute__((address_space(3))) v4i_vs*)(p))
#define GAS32(p)  ((__attribute__((address_space(1))) int*)(p))
#define LAS32(p)  ((__attribute__((address_space(3))) int*)(p))

__device__ __forceinline__ void async_wait0() {
#if __has_builtin(__builtin_amdgcn_s_wait_asynccnt)
  __builtin_amdgcn_s_wait_asynccnt(0);
#else
  asm volatile("s_wait_asynccnt 0x0" ::: "memory");
#endif
}

// ---------------- WMMA fragment helpers (CDNA5 ISA §7.12.2) ----------------
// 16-bit A 16x32 (MxK), wave32, contiguous k order used by this code (not AMD's table order):
//   lanes 0-15: M=lane,     VGPR j holds K pair {2j,2j+1}        (K 0..15)
//   lanes 16-31: M=lane-16, VGPR j holds K pair {16+2j,17+2j}    (K 16..31)
template<bool KLIM16>
__device__ __forceinline__ v16h load_a_frag(const _Float16* base, int ld, int lane) {
  int r  = lane & 15;
  int kb = (lane >> 4) * 16;
  const _Float16* p = base + r * ld;
  v16h a;
#pragma unroll
  for (int j = 0; j < 8; ++j) {
    int k0 = 2 * j + kb;
    if (KLIM16 && k0 >= 16) {            // zero-pad K dim (used for K=HD=16 score GEMM)
      a[2 * j] = (_Float16)0.f; a[2 * j + 1] = (_Float16)0.f;
    } else {
      a[2 * j] = p[k0]; a[2 * j + 1] = p[k0 + 1];
    }
  }
  return a;
}

// 16-bit B 32x16 (KxN), same contiguous k order as A: source stored as [N][K] row-major (i.e. B column-major).
//   lanes 0-15:  N=lane,     VGPR j holds K pair {2j,2j+1}        (K 0..15)
//   lanes 16-31: N=lane-16,  VGPR j holds K pair {16+2j,17+2j}    (K 16..31)
template<bool KLIM16>
__device__ __forceinline__ v16h load_b_frag(const _Float16* base, int ld, int lane) {
  int n  = lane & 15;
  int kb = (lane >> 4) * 16;
  v16h b;
  if (KLIM16 && kb != 0) {
#pragma unroll
    for (int j = 0; j < 16; ++j) b[j] = (_Float16)0.f;
    return b;
  }
  const _Float16* p = base + n * ld + kb;
#pragma unroll
  for (int j = 0; j < 8; ++j) { b[2 * j] = p[2 * j]; b[2 * j + 1] = p[2 * j + 1]; }
  return b;
}

__device__ __forceinline__ v8f wmma16(v16h a, v16h b, v8f c) {
  // D = A(16x32 f16) x B(32x16 f16) + C(16x16 f32)
  return __builtin_amdgcn_wmma_f32_16x16x32_f16(false, a, false, b, (short)0, c, false, false);
}
// f32 C/D 16x16: VGPR i -> M = i + (lane>=16 ? 8 : 0), N = lane&15

// ---------------- Kernel 1: conv + GELU + BN + residual ----------------
__global__ void __launch_bounds__(128)
prep_kernel(const float* __restrict__ x, const float* __restrict__ cw,
            const float* __restrict__ cb, const float* __restrict__ bn_g,
            const float* __restrict__ bn_b, const float* __restrict__ bn_m,
            const float* __restrict__ bn_v,
            float* __restrict__ src32, _Float16* __restrict__ src16) {
  int bc = blockIdx.x;            // b*SEQ + c
  int c  = bc % SEQ;
  int d  = threadIdx.x;           // 0..127
  const float* xr = x + (size_t)bc * DMODEL;
  float acc = cb[c];
#pragma unroll
  for (int k = 0; k < KCONV; ++k) {
    int p = d - 3 + k;            // 'same' even-k padding: left 3, right 4
    float xv = (p >= 0 && p < DMODEL) ? xr[p] : 0.f;
    acc += xv * cw[c * KCONV + k];
  }
  float g = 0.5f * acc * (1.f + erff(acc * 0.70710678118654752f));  // exact GELU
  float s = (g - bn_m[c]) * (bn_g[c] * rsqrtf(bn_v[c] + 1e-5f)) + bn_b[c];
  float out = xr[d] + s;          // residual
  src32[(size_t)bc * DMODEL + d] = out;
  src16[(size_t)bc * DMODEL + d] = (_Float16)out;
}

// ---------------- Kernel 2: weights fp32 -> fp16 ----------------
__global__ void __launch_bounds__(256)
wcvt_kernel(const float* __restrict__ wq, const float* __restrict__ wk,
            const float* __restrict__ wv, const float* __restrict__ wo,
            _Float16* __restrict__ wf16) {
  int i = blockIdx.x * 256 + threadIdx.x;       // 0..65535
  int m = i >> 14, r = i & 16383;
  const float* s = (m == 0) ? wq : (m == 1) ? wk : (m == 2) ? wv : wo;
  wf16[i] = (_Float16)s[r];
}

// ---------------- Kernel 3: QKV projections (per-batch WG, 8 waves) ----------------
__global__ void __launch_bounds__(256)
qkv_kernel(const _Float16* __restrict__ src16, const _Float16* __restrict__ wf16,
           const float* __restrict__ bq, const float* __restrict__ bk,
           const float* __restrict__ bv,
           _Float16* __restrict__ Qg, _Float16* __restrict__ Kg,
           _Float16* __restrict__ Vt) {
  __shared__ _Float16 sSrc[SEQ * DMODEL];       // 28 KB
  int b = blockIdx.x;
  int tid = threadIdx.x, lane = tid & 31, wave = tid >> 5;

  __builtin_prefetch(wf16 + (size_t)tid * 192, 0, 3);   // warm 96KB of QKV weights

#if HAS_ASYNC_LDS
  {
    const char* g = (const char*)(src16 + (size_t)b * SEQ * DMODEL);
    char* l = (char*)sSrc;
    for (int i = tid; i < (SEQ * DMODEL) / 8; i += 256)   // 1792 x 16B chunks
      __builtin_amdgcn_global_load_async_to_lds_b128(GAS128(g + i * 16), LAS128(l + i * 16), 0, 0);
    async_wait0();
  }
#else
  {
    const uint4* gs = (const uint4*)(src16 + (size_t)b * SEQ * DMODEL);
    uint4* ss = (uint4*)sSrc;
    for (int i = tid; i < (SEQ * DMODEL) / 8; i += 256) ss[i] = gs[i];
  }
#endif
  __syncthreads();

#pragma unroll
  for (int o = 0; o < 3; ++o) {
    const _Float16* W   = wf16 + o * DMODEL * DMODEL;   // [N][K] row-major
    const float* bias   = (o == 0) ? bq : (o == 1) ? bk : bv;
    for (int t = wave; t < 56; t += 8) {                // 7 M-tiles x 8 N-tiles
      int mt = t >> 3, nt = t & 7;
      v8f acc = {};
#pragma unroll
      for (int kc = 0; kc < 4; ++kc) {
        v16h a  = load_a_frag<false>(sSrc + (mt * 16) * DMODEL + kc * 32, DMODEL, lane);
        v16h bm = load_b_frag<false>(W + (nt * 16) * DMODEL + kc * 32, DMODEL, lane);
        acc = wmma16(a, bm, acc);
      }
      int col  = (lane & 15) + nt * 16;
      int row0 = mt * 16 + (lane >> 4) * 8;
      float bvv = bias[col];
#pragma unroll
      for (int i = 0; i < 8; ++i) {
        int row = row0 + i;
        _Float16 hv = (_Float16)(acc[i] + bvv);
        if (o == 0)      Qg[((size_t)b * SEQ + row) * DMODEL + col] = hv;
        else if (o == 1) Kg[((size_t)b * SEQ + row) * DMODEL + col] = hv;
        else             Vt[((size_t)b * DMODEL + col) * SEQ + row] = hv; // V^T
      }
    }
  }
}

// ---------------- Kernel 4: attention per (batch, head) ----------------
__global__ void __launch_bounds__(256)
attn_kernel(const _Float16* __restrict__ Qg, const _Float16* __restrict__ Kg,
            const _Float16* __restrict__ Vt, _Float16* __restrict__ Og) {
  __shared__ _Float16 sQ[SEQ * HD];        // 3.5 KB
  __shared__ _Float16 sK[SEQ * HD];        // 3.5 KB
  __shared__ _Float16 sV[HD * DMODEL];     // V^T slice [n][k], k padded 112->128, 4 KB
  __shared__ _Float16 sS[SEQ * DMODEL];    // scores, cols padded 112->128, 28 KB
  int b = blockIdx.x >> 3, h = blockIdx.x & 7;
  int tid = threadIdx.x, lane = tid & 31, wave = tid >> 5;

#if HAS_ASYNC_LDS
  for (int i = tid; i < SEQ * HD / 2; i += 256) {        // 896 dword chunks
    int r = i >> 3, jp = (i & 7) * 2;                    // row, half-pair within row
    size_t go = ((size_t)b * SEQ + r) * DMODEL + h * HD + jp;
    __builtin_amdgcn_global_load_async_to_lds_b32(GAS32(Qg + go), LAS32(sQ + r * HD + jp), 0, 0);
    __builtin_amdgcn_global_load_async_to_lds_b32(GAS32(Kg + go), LAS32(sK + r * HD + jp), 0, 0);
  }
  for (int i = tid; i < HD * (SEQ / 2); i += 256) {      // 16 rows x 56 dwords
    int n = i / (SEQ / 2), kp = (i % (SEQ / 2)) * 2;
    __builtin_amdgcn_global_load_async_to_lds_b32(
        GAS32(Vt + ((size_t)b * DMODEL + h * HD + n) * SEQ + kp),
        LAS32(sV + n * DMODEL + kp), 0, 0);
  }
#else
  for (int i = tid; i < SEQ * HD; i += 256) {
    int r = i >> 4, j = i & 15;
    sQ[i] = Qg[((size_t)b * SEQ + r) * DMODEL + h * HD + j];
    sK[i] = Kg[((size_t)b * SEQ + r) * DMODEL + h * HD + j];
  }
  for (int i = tid; i < HD * SEQ; i += 256) {
    int n = i / SEQ, k = i % SEQ;
    sV[n * DMODEL + k] = Vt[((size_t)b * DMODEL + h * HD + n) * SEQ + k];
  }
#endif
  for (int i = tid; i < HD * 16; i += 256) {             // zero V K-padding
    int n = i >> 4, k = i & 15;
    sV[n * DMODEL + SEQ + k] = (_Float16)0.f;
  }
  for (int i = tid; i < SEQ * 16; i += 256) {            // zero score K-padding
    int r = i >> 4, k = i & 15;
    sS[r * DMODEL + SEQ + k] = (_Float16)0.f;
  }
#if HAS_ASYNC_LDS
  async_wait0();
#endif
  __syncthreads();

  // S = (Q Kh^T) * scale ; K-dim = HD = 16, zero-padded to 32
  const float scale = 0.25f;                             // HD^-0.5
  for (int t = wave; t < 49; t += 8) {
    int mt = t / 7, nt = t % 7;
    v16h a  = load_a_frag<true>(sQ + mt * 16 * HD, HD, lane);
    v16h bm = load_b_frag<true>(sK + nt * 16 * HD, HD, lane);
    v8f acc = {};
    acc = wmma16(a, bm, acc);
    int col  = (lane & 15) + nt * 16;
    int row0 = mt * 16 + (lane >> 4) * 8;
#pragma unroll
    for (int i = 0; i < 8; ++i)
      sS[(row0 + i) * DMODEL + col] = (_Float16)(acc[i] * scale);
  }
  __syncthreads();

  // row softmax over 112 entries
  if (tid < SEQ) {
    _Float16* rp = sS + tid * DMODEL;
    float mx = -1e30f;
    for (int j = 0; j < SEQ; ++j) mx = fmaxf(mx, (float)rp[j]);
    float sum = 0.f;
    for (int j = 0; j < SEQ; ++j) { float e = __expf((float)rp[j] - mx); sum += e; rp[j] = (_Float16)e; }
    float inv = 1.f / sum;
    for (int j = 0; j < SEQ; ++j) rp[j] = (_Float16)((float)rp[j] * inv);
  }
  __syncthreads();

  // O_h = S @ V_h  (K = 112 padded to 128 with zeros in both S cols and sV)
  if (wave < 7) {
    int mt = wave;
    v8f acc = {};
#pragma unroll
    for (int kc = 0; kc < 4; ++kc) {
      v16h a  = load_a_frag<false>(sS + mt * 16 * DMODEL + kc * 32, DMODEL, lane);
      v16h bm = load_b_frag<false>(sV + kc * 32, DMODEL, lane);
      acc = wmma16(a, bm, acc);
    }
    int col  = lane & 15;
    int row0 = mt * 16 + (lane >> 4) * 8;
#pragma unroll
    for (int i = 0; i < 8; ++i)
      Og[((size_t)b * SEQ + row0 + i) * DMODEL + h * HD + col] = (_Float16)acc[i];
  }
}

// ---------------- Kernel 5: out-proj + residual + LayerNorm ----------------
__global__ void __launch_bounds__(256)
final_kernel(const _Float16* __restrict__ Og, const _Float16* __restrict__ wo16,
             const float* __restrict__ bo, const float* __restrict__ src32,
             const float* __restrict__ ln_g, const float* __restrict__ ln_b,
             float* __restrict__ out) {
  __shared__ float sFin[SEQ * DMODEL];     // 56 KB
  int b = blockIdx.x;
  int tid = threadIdx.x, lane = tid & 31, wave = tid >> 5;
  const _Float16* Ob = Og + (size_t)b * SEQ * DMODEL;

  __builtin_prefetch(wo16 + (size_t)tid * 64, 0, 3);     // warm 32KB out-proj weights

  for (int t = wave; t < 56; t += 8) {
    int mt = t >> 3, nt = t & 7;
    v8f acc = {};
#pragma unroll
    for (int kc = 0; kc < 4; ++kc) {
      v16h a  = load_a_frag<false>(Ob + mt * 16 * DMODEL + kc * 32, DMODEL, lane);
      v16h bm = load_b_frag<false>(wo16 + (nt * 16) * DMODEL + kc * 32, DMODEL, lane);
      acc = wmma16(a, bm, acc);
    }
    int col  = (lane & 15) + nt * 16;
    int row0 = mt * 16 + (lane >> 4) * 8;
    float bvv = bo[col];
#pragma unroll
    for (int i = 0; i < 8; ++i) {
      int row = row0 + i;
      sFin[row * DMODEL + col] =
          acc[i] + bvv + src32[((size_t)b * SEQ + row) * DMODEL + col];
    }
  }
  __syncthreads();

  if (tid < SEQ) {
    float* rp = sFin + tid * DMODEL;
    float mu = 0.f;
    for (int j = 0; j < DMODEL; ++j) mu += rp[j];
    mu *= (1.f / DMODEL);
    float var = 0.f;
    for (int j = 0; j < DMODEL; ++j) { float d = rp[j] - mu; var += d * d; }
    var *= (1.f / DMODEL);
    float rs = rsqrtf(var + 1e-5f);
    float* orow = out + ((size_t)b * SEQ + tid) * DMODEL;
    for (int j = 0; j < DMODEL; ++j)
      orow[j] = (rp[j] - mu) * rs * ln_g[j] + ln_b[j];
  }
}

// ---------------- Host launcher ----------------
extern "C" void kernel_launch(void* const* d_in, const int* in_sizes, int n_in,
                              void* d_out, int out_size, void* d_ws, size_t ws_size,
                              hipStream_t stream) {
  const float* x      = (const float*)d_in[0];
  const float* conv_w = (const float*)d_in[1];
  const float* conv_b = (const float*)d_in[2];
  const float* bn_g   = (const float*)d_in[3];
  const float* bn_b   = (const float*)d_in[4];
  const float* bn_m   = (const float*)d_in[5];
  const float* bn_v   = (const float*)d_in[6];
  const float* wq     = (const float*)d_in[7];
  const float* bq     = (const float*)d_in[8];
  const float* wk     = (const float*)d_in[9];
  const float* bk     = (const float*)d_in[10];
  const float* wv     = (const float*)d_in[11];
  const float* bv     = (const float*)d_in[12];
  const float* wo     = (const float*)d_in[13];
  const float* bo     = (const float*)d_in[14];
  const float* ln_g   = (const float*)d_in[15];
  const float* ln_b   = (const float*)d_in[16];
  float* out = (float*)d_out;

  const size_t NE = (size_t)BATCH * SEQ * DMODEL;  // 7,340,032 elements
  char* ws = (char*)d_ws;
  size_t off = 0;
  auto carve = [&](size_t bytes) -> char* {
    char* p = ws + off;
    off += (bytes + 255) & ~(size_t)255;
    return p;
  };
  float*    src32 = (float*)   carve(NE * sizeof(float));
  _Float16* src16 = (_Float16*)carve(NE * sizeof(_Float16));
  _Float16* wf16  = (_Float16*)carve(4 * DMODEL * DMODEL * sizeof(_Float16));
  _Float16* Qg    = (_Float16*)carve(NE * sizeof(_Float16));
  _Float16* Kg    = (_Float16*)carve(NE * sizeof(_Float16));
  _Float16* Vt    = (_Float16*)carve(NE * sizeof(_Float16));
  _Float16* Og    = (_Float16*)carve(NE * sizeof(_Float16));
  (void)ws_size; (void)n_in; (void)in_sizes; (void)out_size;

  prep_kernel<<<BATCH * SEQ, 128, 0, stream>>>(x, conv_w, conv_b, bn_g, bn_b,
                                               bn_m, bn_v, src32, src16);
  wcvt_kernel<<<(4 * DMODEL * DMODEL) / 256, 256, 0, stream>>>(wq, wk, wv, wo, wf16);
  qkv_kernel<<<BATCH, 256, 0, stream>>>(src16, wf16, bq, bk, bv, Qg, Kg, Vt);
  attn_kernel<<<BATCH * NHEADS, 256, 0, stream>>>(Qg, Kg, Vt, Og);
  final_kernel<<<BATCH, 256, 0, stream>>>(Og, wf16 + 3 * DMODEL * DMODEL, bo,
                                          src32, ln_g, ln_b, out);
}
